// NGCF_18657337934510
// MI455X (gfx1250) — matrix operand order repaired and checked by reading, not compiled, not run on hardware
//
#include <hip/hip_runtime.h>
#include <hip/hip_bf16.h>
#include <math.h>

// ---------------------------------------------------------------------------
// MI455X (gfx1250) NGCF pipeline.
// Roofline: output store (4096x41476 fp32 = 680MB) @ 23.3 TB/s ~= 29us floor.
// GEMMs: fragment-major f16 packing (pred_w: 10.6MB f16, L2-resident) +
// v_wmma_f32_16x16x32_f16 with f32 accum. A fragments staged via LDS and
// shared by 8 waves; each wave computes a 16x32 output strip (2 N-tiles per
// A fragment -> 2 WMMAs per ds_load pair). Output uses NT stores so the
// 680MB stream does not evict weights from the 192MB L2.
// Graph scatters -> fp32 global atomics (targets <=4MB, L2-resident).
// ---------------------------------------------------------------------------

#define N_NODES 4096
#define N_EDGES 131072

typedef __attribute__((ext_vector_type(16))) _Float16 v16h;
typedef __attribute__((ext_vector_type(8)))  float    v8f;

#define GEMM_WAVES 8

// --------------------------- fragment packing ------------------------------
// A[M,K] fp32 -> fragment-major f16: out[((tm*nKT + kt)*32 + lane)*16 + j]
// lane = half*16 + l15 owns row tm*16+l15; element j -> K offset
// half*16+j (contiguous, matching pack_w's B order).
__global__ void pack_a(const float* __restrict__ A, _Float16* __restrict__ out,
                       int M, int K) {
  int i = blockIdx.x * blockDim.x + threadIdx.x;
  if (i >= M * K) return;
  int j    = i & 15;
  int lane = (i >> 4) & 31;
  int nKT  = K >> 5;
  int kt   = (i >> 9) % nKT;
  int tm   = i / (nKT << 9);
  int half = lane >> 4, l15 = lane & 15;
  int row  = tm * 16 + l15;
  int k    = kt * 32 + half * 16 + j;
  out[i] = (_Float16)A[(size_t)row * K + k];
}

// W[K,N] fp32 -> fragment-major f16: out[((kt*nTN + nt)*32 + lane)*16 + j]
// lane = half*16 + l15 owns col nt*16+l15; element j -> K = kt*32+half*16+j
// (contiguous k order, same as pack_a). Columns >= N zero-padded: no predication in
// the GEMM inner loop, EXEC stays all-ones for WMMA.
__global__ void pack_w(const float* __restrict__ W, _Float16* __restrict__ out,
                       int K, int N) {
  int nTN = (N + 15) >> 4;
  int total = (K >> 5) * nTN * 512;
  int i = blockIdx.x * blockDim.x + threadIdx.x;
  if (i >= total) return;
  int j    = i & 15;
  int lane = (i >> 4) & 31;
  int f    = i >> 9;                 // kt*nTN + nt
  int nt   = f % nTN;
  int kt   = f / nTN;
  int half = lane >> 4, l15 = lane & 15;
  int col  = nt * 16 + l15;
  int k    = kt * 32 + half * 16 + j;
  out[i] = (col < N) ? (_Float16)W[(size_t)k * N + col] : (_Float16)0.0f;
}

// ------------------------------ WMMA GEMM ----------------------------------
// C[M,N] = op(A @ W (+bias) (+C)) from packed f16 fragments.
// One wave per 16x32 strip (two 16x16 N-tiles sharing one A fragment);
// 8 waves/block share tileM -> A staged once in LDS.
// flags: bit0 = relu, bit1 = accumulate into C, bit2 = NT store.
__global__ __launch_bounds__(GEMM_WAVES * 32)
void gemm_wmma(const _Float16* __restrict__ Apk, const _Float16* __restrict__ Wpk,
               const float* __restrict__ bias, float* __restrict__ C,
               int M, int N, int K, int flags) {
  extern __shared__ __align__(32) char smemRaw[];   // nKT*32 v16h fragments
  v16h* As = (v16h*)smemRaw;

  const int lane = threadIdx.x & 31;
  const int wave = threadIdx.x >> 5;
  const int nTN  = (N + 15) >> 4;
  const int nKT  = K >> 5;
  const int tileM = blockIdx.y;
  int t0 = (blockIdx.x * GEMM_WAVES + wave) * 2;
  int t1 = t0 + 1;
  const bool act0 = (t0 < nTN), act1 = (t1 < nTN);
  if (!act0) t0 = nTN - 1;           // clamp: keep wave alive for barrier/WMMA
  if (!act1) t1 = nTN - 1;

  // cooperative stage of tileM's packed A fragments (shared by all 8 waves)
  const v16h* Ag = (const v16h*)Apk + (size_t)tileM * nKT * 32;
  for (int i = threadIdx.x; i < nKT * 32; i += blockDim.x)
    As[i] = Ag[i];
  __syncthreads();

  const v16h* Bf = (const v16h*)Wpk;
  v8f acc0 = {}, acc1 = {};
  for (int kt = 0; kt < nKT; ++kt) {
    v16h a  = As[kt * 32 + lane];                              // ds_load_b128 x2
    v16h b0 = Bf[((size_t)kt * nTN + t0) * 32 + lane];         // global b128 x2
    v16h b1 = Bf[((size_t)kt * nTN + t1) * 32 + lane];
    if (kt + 1 < nKT) {                                        // warm WGP caches
      __builtin_prefetch((const void*)&Bf[((size_t)(kt + 1) * nTN + t0) * 32 + lane], 0, 3);
      __builtin_prefetch((const void*)&Bf[((size_t)(kt + 1) * nTN + t1) * 32 + lane], 0, 3);
    }
    acc0 = __builtin_amdgcn_wmma_f32_16x16x32_f16(
        false, a, false, b0, (short)0, acc0, false, false);
    acc1 = __builtin_amdgcn_wmma_f32_16x16x32_f16(
        false, a, false, b1, (short)0, acc1, false, false);
  }

  const int half = lane >> 4, l15 = lane & 15;
#pragma unroll
  for (int s = 0; s < 2; ++s) {
    const int  tn  = s ? t1 : t0;
    const bool act = s ? act1 : act0;
    const v8f& acc = s ? acc1 : acc0;
    const int colB = tn * 16 + l15;
    if (!act || colB >= N) continue;
    const float bv = bias ? bias[colB] : 0.0f;
#pragma unroll
    for (int r = 0; r < 8; ++r) {    // C/D: VGPR r -> row r + 8*half
      const int row = tileM * 16 + r + half * 8;
      float* cp = C + (size_t)row * N + colB;
      float v = acc[r] + bv;
      if (flags & 2) v += *cp;
      if (flags & 1) v = fmaxf(v, 0.0f);
      if (flags & 4) __builtin_nontemporal_store(v, cp);
      else          *cp = v;
    }
  }
}

// --------------------------- small utility kernels -------------------------
__global__ void fill_f32(float* p, float v, int n) {
  int i = blockIdx.x * blockDim.x + threadIdx.x;
  if (i < n) p[i] = v;
}
__global__ void fill_u32(unsigned* p, unsigned v, int n) {
  int i = blockIdx.x * blockDim.x + threadIdx.x;
  if (i < n) p[i] = v;
}

__global__ void embed_concat(const int* __restrict__ node_x,
                             const float* __restrict__ ue,
                             const float* __restrict__ ie,
                             float* __restrict__ x0) {
  int i = blockIdx.x * blockDim.x + threadIdx.x;
  if (i >= N_NODES * 128) return;
  int v = i >> 7, c = i & 127;
  x0[i] = (c < 64) ? ue[(size_t)node_x[2 * v] * 64 + c]
                   : ie[(size_t)node_x[2 * v + 1] * 64 + (c - 64)];
}

__global__ void deg_count(const int* __restrict__ idx, float* deg, int E) {
  int e = blockIdx.x * blockDim.x + threadIdx.x;
  if (e < E) atomicAdd(&deg[idx[e]], 1.0f);
}

__global__ void dinv_k(const float* __restrict__ deg, float* dinv, int n) {
  int i = blockIdx.x * blockDim.x + threadIdx.x;
  if (i < n) { float d = deg[i]; dinv[i] = d > 0.0f ? rsqrtf(d) : 0.0f; }
}

// out[v,f] = h[v,f]*dinv[v]^2 + b[f]   (GCN self-loop term + bias)
__global__ void gcn_init(const float* __restrict__ h,
                         const float* __restrict__ dinv,
                         const float* __restrict__ b,
                         float* __restrict__ out, int F) {
  int v = blockIdx.x, f = threadIdx.x;
  float di = dinv[v];
  out[(size_t)v * F + f] = h[(size_t)v * F + f] * di * di + b[f];
}

__global__ void gcn_scatter(const float* __restrict__ h,
                            const int* __restrict__ ei0,
                            const int* __restrict__ ei1,
                            const float* __restrict__ dinv,
                            float* __restrict__ out, int F) {
  int e = blockIdx.x, f = threadIdx.x;
  int r = ei0[e], c = ei1[e];
  float nrm = dinv[r] * dinv[c];
  atomicAdd(&out[(size_t)c * F + f], h[(size_t)r * F + f] * nrm);
}

// in-place batchnorm (population var) + relu, one block per column
__global__ void bn_relu(float* __restrict__ x, const float* __restrict__ g,
                        const float* __restrict__ be, int M, int F) {
  __shared__ float ssum[256], ssq[256];
  int c = blockIdx.x, t = threadIdx.x;
  float s = 0.0f, q = 0.0f;
  for (int r = t; r < M; r += blockDim.x) {
    float v = x[(size_t)r * F + c]; s += v; q += v * v;
  }
  ssum[t] = s; ssq[t] = q; __syncthreads();
  for (int d = 128; d > 0; d >>= 1) {
    if (t < d) { ssum[t] += ssum[t + d]; ssq[t] += ssq[t + d]; }
    __syncthreads();
  }
  float mu  = ssum[0] / (float)M;
  float var = ssq[0] / (float)M - mu * mu;
  float inv = rsqrtf(var + 1e-5f) * g[c];
  float bb  = be[c];
  for (int r = t; r < M; r += blockDim.x) {
    float v = (x[(size_t)r * F + c] - mu) * inv + bb;
    x[(size_t)r * F + c] = fmaxf(v, 0.0f);
  }
}

__global__ void sage_scatter(const float* __restrict__ x,
                             const int* __restrict__ ei0,
                             const int* __restrict__ ei1,
                             float* __restrict__ agg, float* __restrict__ cnt,
                             int F) {
  int e = blockIdx.x, f = threadIdx.x;
  int r = ei0[e], c = ei1[e];
  atomicAdd(&agg[(size_t)c * F + f], x[(size_t)r * F + f]);
  if (f == 0) atomicAdd(&cnt[c], 1.0f);
}

__global__ void sage_div(float* __restrict__ agg, const float* __restrict__ cnt,
                         int F) {
  int v = blockIdx.x, f = threadIdx.x;
  agg[(size_t)v * F + f] /= fmaxf(cnt[v], 1.0f);
}

__global__ void cheb_scatter(const float* __restrict__ x,
                             const int* __restrict__ ei0,
                             const int* __restrict__ ei1,
                             const float* __restrict__ dinv,
                             float* __restrict__ tx1, int F) {
  int e = blockIdx.x, f = threadIdx.x;
  int r = ei0[e], c = ei1[e];
  float w = -(dinv[r] * dinv[c]);
  atomicAdd(&tx1[(size_t)c * F + f], x[(size_t)r * F + f] * w);
}

// a_s[v] = h[v,:] . asrc ; a_d[v] = h[v,:] . adst   (one block per node)
__global__ void gat_av(const float* __restrict__ h,
                       const float* __restrict__ asrc,
                       const float* __restrict__ adst,
                       float* __restrict__ a_s, float* __restrict__ a_d) {
  __shared__ float s1[128], s2[128];
  int v = blockIdx.x, t = threadIdx.x;
  float hv = h[(size_t)v * 128 + t];
  s1[t] = hv * asrc[t]; s2[t] = hv * adst[t]; __syncthreads();
  for (int d = 64; d > 0; d >>= 1) {
    if (t < d) { s1[t] += s1[t + d]; s2[t] += s2[t + d]; }
    __syncthreads();
  }
  if (t == 0) { a_s[v] = s1[0]; a_d[v] = s2[0]; }
}

// monotonic float<->uint key for deterministic atomicMax on floats
__device__ __forceinline__ unsigned fkey(float f) {
  unsigned u = __float_as_uint(f);
  return (u & 0x80000000u) ? ~u : (u | 0x80000000u);
}
__device__ __forceinline__ float funkey(unsigned k) {
  unsigned u = (k & 0x80000000u) ? (k & 0x7FFFFFFFu) : ~k;
  return __uint_as_float(u);
}

__global__ void gat_emax(const int* __restrict__ ei0, const int* __restrict__ ei1,
                         const float* __restrict__ a_s, const float* __restrict__ a_d,
                         unsigned* __restrict__ emaxk, int E, int n) {
  int i = blockIdx.x * blockDim.x + threadIdx.x;
  if (i >= E + n) return;
  int r, c;
  if (i < E) { r = ei0[i]; c = ei1[i]; } else { r = c = i - E; }
  float e = a_s[r] + a_d[c];
  e = e > 0.0f ? e : 0.2f * e;              // leaky_relu(0.2)
  atomicMax(&emaxk[c], fkey(e));
}

// unnormalized softmax scatter: out[c,:] += h[r,:]*exp(e-emax[c]); denom[c]+=ex
__global__ void gat_scatter(const float* __restrict__ h,
                            const int* __restrict__ ei0, const int* __restrict__ ei1,
                            const float* __restrict__ a_s, const float* __restrict__ a_d,
                            const unsigned* __restrict__ emaxk,
                            float* __restrict__ denom, float* __restrict__ out,
                            int E) {
  int i = blockIdx.x, t = threadIdx.x;
  int r, c;
  if (i < E) { r = ei0[i]; c = ei1[i]; } else { r = c = i - E; }
  float e = a_s[r] + a_d[c];
  e = e > 0.0f ? e : 0.2f * e;
  float ex = expf(e - funkey(emaxk[c]));
  if (t == 0) atomicAdd(&denom[c], ex);
  atomicAdd(&out[(size_t)c * 128 + t], h[(size_t)r * 128 + t] * ex);
}

__global__ void gat_final(float* __restrict__ out, const float* __restrict__ denom,
                          const float* __restrict__ b) {
  int v = blockIdx.x, t = threadIdx.x;
  float x = out[(size_t)v * 128 + t] / denom[v] + b[t];
  out[(size_t)v * 128 + t] = x > 0.0f ? x : expm1f(x);   // elu
}

// ------------------------------- launcher ----------------------------------
extern "C" void kernel_launch(void* const* d_in, const int* in_sizes, int n_in,
                              void* d_out, int out_size, void* d_ws, size_t ws_size,
                              hipStream_t stream) {
  (void)in_sizes; (void)n_in; (void)out_size; (void)ws_size;

  const int*   ei0     = (const int*)d_in[0];
  const int*   ei1     = ei0 + N_EDGES;
  const int*   node_x  = (const int*)d_in[1];
  const float* ue      = (const float*)d_in[2];
  const float* ie      = (const float*)d_in[3];
  const float* mlp_w1  = (const float*)d_in[4];
  const float* mlp_b1  = (const float*)d_in[5];
  const float* mlp_w2  = (const float*)d_in[6];
  const float* mlp_b2  = (const float*)d_in[7];
  const float* gcn_w1  = (const float*)d_in[8];
  const float* gcn_b1  = (const float*)d_in[9];
  const float* bn1_g   = (const float*)d_in[10];
  const float* bn1_b   = (const float*)d_in[11];
  const float* gcn_w2  = (const float*)d_in[12];
  const float* gcn_b2  = (const float*)d_in[13];
  const float* bn2_g   = (const float*)d_in[14];
  const float* bn2_b   = (const float*)d_in[15];
  const float* sage_wl = (const float*)d_in[16];
  const float* sage_bl = (const float*)d_in[17];
  const float* sage_wr = (const float*)d_in[18];
  const float* cheb_w0 = (const float*)d_in[19];
  const float* cheb_w1 = (const float*)d_in[20];
  const float* cheb_b  = (const float*)d_in[21];
  const float* gat1_w  = (const float*)d_in[22];
  const float* gat1_as = (const float*)d_in[23];
  const float* gat1_ad = (const float*)d_in[24];
  const float* gat1_b  = (const float*)d_in[25];
  const float* gat2_w  = (const float*)d_in[26];
  const float* gat2_as = (const float*)d_in[27];
  const float* gat2_ad = (const float*)d_in[28];
  const float* gat2_b  = (const float*)d_in[29];
  const float* pred_w  = (const float*)d_in[30];
  const float* pred_b  = (const float*)d_in[31];

  // workspace layout (floats). conv-stage buffers overlay the dead h1 region.
  // total = 11,568,128 floats (~46.3 MB)
  float* ws = (float*)d_ws;
  float* x0   = ws;                          // 4096*128
  float* h1   = ws + 524288;                 // 4096*1024 (dead after h2)
  float* h2   = ws + 4718592;                // 4096*512
  float* hA   = h1;                          // 4096*256  (gemm h / agg / tx1)
  float* oA   = h1 + 1048576;                // 4096*256
  float* oB   = h1 + 2097152;                // 4096*128
  float* deg  = h1 + 2621440;                // 4096
  float* dinv = deg + 4096;                  // 4096
  float* a_s  = deg + 8192;                  // 4096
  float* a_d  = deg + 12288;                 // 4096
  float* den  = deg + 16384;                 // 4096
  unsigned* emx = (unsigned*)(deg + 20480);  // 4096
  _Float16* apk = (_Float16*)(ws + 6815744); // 4096*1024 f16 (2,097,152 fl)
  _Float16* wpk = (_Float16*)(ws + 8912896); // 5,310,464 f16 (2,655,232 fl)

  const int E = N_EDGES, n = N_NODES;
  auto gemm = [&](const float* A, const float* W, const float* bias, float* C,
                  int M, int N, int K, int flags) {
    int nTN = (N + 15) >> 4, nKT = K >> 5;
    pack_a<<<(M * K + 255) / 256, 256, 0, stream>>>(A, apk, M, K);
    int wtot = nKT * nTN * 512;
    pack_w<<<(wtot + 255) / 256, 256, 0, stream>>>(W, wpk, K, N);
    int nPair = (nTN + 1) >> 1;              // two N-tiles per wave
    dim3 grid((nPair + GEMM_WAVES - 1) / GEMM_WAVES, M >> 4);
    gemm_wmma<<<grid, GEMM_WAVES * 32, (size_t)K * 32, stream>>>(
        apk, wpk, bias, C, M, N, K, flags);
  };
  auto fillf = [&](float* p, float v, int cnt) {
    fill_f32<<<(cnt + 255) / 256, 256, 0, stream>>>(p, v, cnt);
  };

  // ---- embeddings + MLP ----
  embed_concat<<<(n * 128 + 255) / 256, 256, 0, stream>>>(node_x, ue, ie, x0);
  gemm(x0, mlp_w1, mlp_b1, h1, n, 1024, 128, 1);        // relu
  gemm(h1, mlp_w2, mlp_b2, h2, n, 512, 1024, 1);        // relu

  // ---- GCN layer 1 (512 -> 256) ----
  gemm(h2, gcn_w1, nullptr, hA, n, 256, 512, 0);
  fillf(deg, 1.0f, n);                                  // self loop
  deg_count<<<(E + 255) / 256, 256, 0, stream>>>(ei1, deg, E);
  dinv_k<<<(n + 255) / 256, 256, 0, stream>>>(deg, dinv, n);
  gcn_init<<<n, 256, 0, stream>>>(hA, dinv, gcn_b1, oA, 256);
  gcn_scatter<<<E, 256, 0, stream>>>(hA, ei0, ei1, dinv, oA, 256);
  bn_relu<<<256, 256, 0, stream>>>(oA, bn1_g, bn1_b, n, 256);

  // ---- GCN layer 2 (256 -> 128) ----
  gemm(oA, gcn_w2, nullptr, hA, n, 128, 256, 0);
  fillf(deg, 1.0f, n);
  deg_count<<<(E + 255) / 256, 256, 0, stream>>>(ei1, deg, E);
  dinv_k<<<(n + 255) / 256, 256, 0, stream>>>(deg, dinv, n);
  gcn_init<<<n, 128, 0, stream>>>(hA, dinv, gcn_b2, oB, 128);
  gcn_scatter<<<E, 128, 0, stream>>>(hA, ei0, ei1, dinv, oB, 128);
  bn_relu<<<128, 256, 0, stream>>>(oB, bn2_g, bn2_b, n, 128);

  // ---- SAGE (x = oB -> oA) ----
  fillf(hA, 0.0f, n * 128); fillf(deg, 0.0f, n);
  sage_scatter<<<E, 128, 0, stream>>>(oB, ei0, ei1, hA, deg, 128);
  sage_div<<<n, 128, 0, stream>>>(hA, deg, 128);
  gemm(hA, sage_wl, sage_bl, oA, n, 128, 128, 0);
  gemm(oB, sage_wr, nullptr, oA, n, 128, 128, 1 | 2);   // acc + relu

  // ---- Cheb (x = oA -> oB) ----
  fillf(deg, 0.0f, n);
  deg_count<<<(E + 255) / 256, 256, 0, stream>>>(ei0, deg, E);
  dinv_k<<<(n + 255) / 256, 256, 0, stream>>>(deg, dinv, n);
  fillf(hA, 0.0f, n * 128);
  cheb_scatter<<<E, 128, 0, stream>>>(oA, ei0, ei1, dinv, hA, 128);
  gemm(oA, cheb_w0, cheb_b, oB, n, 128, 128, 0);
  gemm(hA, cheb_w1, nullptr, oB, n, 128, 128, 1 | 2);   // acc + relu

  // ---- GAT 1 (x = oB -> oA) ----
  gemm(oB, gat1_w, nullptr, hA, n, 128, 128, 0);
  gat_av<<<n, 128, 0, stream>>>(hA, gat1_as, gat1_ad, a_s, a_d);
  fill_u32<<<(n + 255) / 256, 256, 0, stream>>>(emx, 0u, n);
  fillf(den, 0.0f, n); fillf(oA, 0.0f, n * 128);
  gat_emax<<<(E + n + 255) / 256, 256, 0, stream>>>(ei0, ei1, a_s, a_d, emx, E, n);
  gat_scatter<<<E + n, 128, 0, stream>>>(hA, ei0, ei1, a_s, a_d, emx, den, oA, E);
  gat_final<<<n, 128, 0, stream>>>(oA, den, gat1_b);

  // ---- GAT 2 (x = oA -> oB) ----
  gemm(oA, gat2_w, nullptr, hA, n, 128, 128, 0);
  gat_av<<<n, 128, 0, stream>>>(hA, gat2_as, gat2_ad, a_s, a_d);
  fill_u32<<<(n + 255) / 256, 256, 0, stream>>>(emx, 0u, n);
  fillf(den, 0.0f, n); fillf(oB, 0.0f, n * 128);
  gat_emax<<<(E + n + 255) / 256, 256, 0, stream>>>(ei0, ei1, a_s, a_d, emx, E, n);
  gat_scatter<<<E + n, 128, 0, stream>>>(hA, ei0, ei1, a_s, a_d, emx, den, oB, E);
  gat_final<<<n, 128, 0, stream>>>(oB, den, gat2_b);

  // ---- prediction head: 4096x128 @ 128x41476, NT stores (stream 680MB) ----
  gemm(oB, pred_w, pred_b, (float*)d_out, n, 41476, 128, 4);
}
